// D_SS2D_90984587198516
// MI455X (gfx1250) — matrix operand order repaired and checked by reading, not compiled, not run on hardware
//
#include <hip/hip_runtime.h>
#include <hip/hip_bf16.h>
#include <math.h>

// ---------------- problem constants (from reference) ----------------
#define BB       2
#define HH       64
#define WW       64
#define LL       (HH * WW)          // 4096
#define DMODEL   96
#define DINNER   192                // 2 * DMODEL
#define DSTATE   16
#define DTRANK   6                  // ceil(96/16)
#define KK       2
#define CDIM     (DTRANK + 2 * DSTATE)  // 38
#define CPAD     48                 // 38 padded to 3 row-tiles
#define MROWS    (BB * LL)          // 8192
#define XCHUNK   64                 // xdbl column chunk

typedef __attribute__((ext_vector_type(16))) _Float16 v16h;
typedef __attribute__((ext_vector_type(8)))  _Float16 v8h;
typedef __attribute__((ext_vector_type(8)))  float    v8f;
typedef __attribute__((ext_vector_type(4)))  int      v4i;

#if __has_builtin(__builtin_amdgcn_global_load_async_to_lds_b128)
#define HAS_ASYNC_LDS 1
#else
#define HAS_ASYNC_LDS 0
#endif

// 16-byte global -> LDS copy (async when the gfx1250 builtin is available)
__device__ __forceinline__ void copy16(const _Float16* g, _Float16* l) {
#if HAS_ASYNC_LDS
  // builtin signature (from hipcc diagnostic): (v4i* global, v4i* lds, imm, imm)
  __builtin_amdgcn_global_load_async_to_lds_b128((v4i*)g, (v4i*)l, 0, 0);
#else
  *(v8h*)l = *(const v8h*)g;
#endif
}

__device__ __forceinline__ void async_copies_done() {
#if HAS_ASYNC_LDS
#if __has_builtin(__builtin_amdgcn_s_wait_asynccnt)
  __builtin_amdgcn_s_wait_asynccnt(0);
#else
  asm volatile("s_wait_asynccnt 0" ::: "memory");
#endif
#endif
  __syncthreads();
}

// ---------------- WMMA fragment loaders from global f32 (unguarded) ---------
// A 16x32 (MxK) f16: lanes 0-15 -> rows, K 0..15; lanes 16-31 K 16..31
__device__ __forceinline__ v16h load_fragA(const float* __restrict__ p,
                                           long rs, long cs,
                                           int r0, int k0, int lane) {
  int r  = r0 + (lane & 15);
  int kh = (lane >> 4) * 16;
  v16h a;
#pragma unroll
  for (int i = 0; i < 8; ++i) {
    int kp = k0 + kh + 2 * i;
    a[2 * i]     = (_Float16)p[(long)r * rs + (long)kp * cs];
    a[2 * i + 1] = (_Float16)p[(long)r * rs + (long)(kp + 1) * cs];
  }
  return a;
}

// B 32x16 (KxN) f16: lane = column; lanes 0-15 K=0..15, lanes 16-31 K=16..31
__device__ __forceinline__ v16h load_fragB(const float* __restrict__ p,
                                           long ks, long ns,
                                           int k0, int n0, int lane) {
  int n  = n0 + (lane & 15);
  int kh = (lane >> 4) * 16;
  v16h b;
#pragma unroll
  for (int j = 0; j < 8; ++j) {
    int kp = k0 + kh + 2 * j;
    b[2 * j]     = (_Float16)p[(long)kp * ks + (long)n * ns];
    b[2 * j + 1] = (_Float16)p[(long)(kp + 1) * ks + (long)n * ns];
  }
  return b;
}

__device__ __forceinline__ float silu_f(float x) {
  return x / (1.f + __expf(-x));
}

// ---------------- K0: diagonal order + inverse ----------------
__global__ void k_order(int* __restrict__ order, int* __restrict__ inv) {
  int p = blockIdx.x * blockDim.x + threadIdx.x;
  if (p >= LL) return;
  int i, j;
  if (p < HH) {
    i = p; j = p;
  } else if (p < HH + (HH * (HH - 1)) / 2) {      // upper triangle, row-major
    int q = p - HH;
    i = 0;
    while (q >= (HH - 1) - i) { q -= (HH - 1) - i; ++i; }
    j = i + 1 + q;
  } else {                                        // lower triangle, row-major
    int q = p - HH - (HH * (HH - 1)) / 2;
    i = 1;
    while (q >= i) { q -= i; ++i; }
    j = q;
  }
  int loc = i * HH + j;
  order[p] = loc;
  inv[loc] = p;
}

// ---------------- K1: input projection  xz = x @ w_in^T (WMMA) ----------------
__global__ void k_in_proj(const float* __restrict__ x,
                          const float* __restrict__ w_in,
                          float* __restrict__ xz) {
  const int N = 2 * DINNER, Kd = DMODEL;
  int wave = (blockIdx.x * blockDim.x + threadIdx.x) >> 5;
  int lane = threadIdx.x & 31;
  int tn = wave % (N / 16);   // 24
  int tm = wave / (N / 16);   // 512
  v8f c = {};
#pragma unroll
  for (int k0 = 0; k0 < Kd; k0 += 32) {
    v16h a = load_fragA(x, Kd, 1, tm * 16, k0, lane);
    v16h b = load_fragB(w_in, 1, Kd, k0, tn * 16, lane);
    c = __builtin_amdgcn_wmma_f32_16x16x32_f16(false, a, false, b,
                                               (short)0, c, false, false);
  }
  int row0 = tm * 16 + (lane >> 4) * 8;
  int col  = tn * 16 + (lane & 15);
#pragma unroll
  for (int v = 0; v < 8; ++v)
    xz[(long)(row0 + v) * N + col] = c[v];
}

// ---------------- K2: depthwise 3x3 conv + bias + SiLU ----------------
__global__ void k_conv(const float* __restrict__ xz,
                       const float* __restrict__ conv_w,
                       const float* __restrict__ conv_b,
                       float* __restrict__ xc) {
  long idx = (long)blockIdx.x * blockDim.x + threadIdx.x;
  if (idx >= (long)BB * DINNER * LL) return;
  int l = (int)(idx & (LL - 1));
  int c = (int)((idx >> 12) % DINNER);
  int b = (int)(idx / ((long)DINNER * LL));
  int h = l >> 6, w = l & 63;
  float acc = conv_b[c];
#pragma unroll
  for (int dy = 0; dy < 3; ++dy) {
#pragma unroll
    for (int dx = 0; dx < 3; ++dx) {
      int hh = h + dy - 1, ww = w + dx - 1;
      if (hh >= 0 && hh < HH && ww >= 0 && ww < WW) {
        acc += conv_w[c * 9 + dy * 3 + dx] *
               xz[((long)(b * LL + hh * WW + ww)) * (2 * DINNER) + c];
      }
    }
  }
  xc[((long)b * DINNER + c) * LL + l] = silu_f(acc);
}

// ---------------- K3: build xs (f32 scan layout + blocked f16 GEMM layout) ---
// xs:    [bk][d][l]                         (f32, for the scan)
// xsh_t: [bk][chunk=l/64][l%64][d]          (f16, contiguous 64x192 B panels)
__global__ void k_build_xs(const float* __restrict__ xc,
                           const int* __restrict__ order,
                           float* __restrict__ xs,
                           _Float16* __restrict__ xsh_t) {
  long idx = (long)blockIdx.x * blockDim.x + threadIdx.x;
  if (idx >= (long)BB * KK * DINNER * LL) return;
  int l  = (int)(idx & (LL - 1));
  int d  = (int)((idx >> 12) % DINNER);
  int bk = (int)(idx / ((long)DINNER * LL));
  int b = bk >> 1, k = bk & 1;
  int pos = (k == 0) ? order[l] : order[LL - 1 - l];
  float val = xc[((long)b * DINNER + d) * LL + pos];
  xs[idx] = val;
  int cb = l >> 6, lp = l & 63;
  xsh_t[(((long)bk * (LL / XCHUNK) + cb) * XCHUNK + lp) * DINNER + d] =
      (_Float16)val;
}

// ---------------- K3b: pack x_proj_w -> zero-padded 48x192 f16 panels -------
__global__ void k_pack_aproj(const float* __restrict__ x_proj_w,
                             _Float16* __restrict__ aph) {
  int idx = blockIdx.x * blockDim.x + threadIdx.x;
  if (idx >= KK * CPAD * DINNER) return;
  int d = idx % DINNER;
  int m = (idx / DINNER) % CPAD;
  int k = idx / (CPAD * DINNER);
  float v = (m < CDIM) ? x_proj_w[((long)k * CDIM + m) * DINNER + d] : 0.f;
  aph[idx] = (_Float16)v;
}

// ---------------- K4: x_dbl = x_proj_w[k] @ xs[b,k]  (LDS-staged WMMA) -------
// block: 128 threads (4 waves); one (bk, 64-col chunk) per block.
__global__ void __launch_bounds__(128)
k_xdbl(const _Float16* __restrict__ xsh_t,
       const _Float16* __restrict__ aph,
       float* __restrict__ x_dbl) {
  __shared__ _Float16 sA[CPAD * DINNER];     // 48x192 f16 = 18 KB
  __shared__ _Float16 sB[XCHUNK * DINNER];   // 64x192 f16 = 24 KB
  int bk = blockIdx.x >> 6;        // 4
  int cc = blockIdx.x & 63;        // 64 chunks of 64 columns
  int k  = bk & 1;
  int tid = threadIdx.x;

  const _Float16* gA = aph + (long)k * CPAD * DINNER;
  const _Float16* gB = xsh_t + ((long)bk * (LL / XCHUNK) + cc) * XCHUNK * DINNER;
  for (int i = tid; i < (CPAD * DINNER) / 8; i += 128)
    copy16(gA + (long)i * 8, &sA[i * 8]);
  for (int i = tid; i < (XCHUNK * DINNER) / 8; i += 128)
    copy16(gB + (long)i * 8, &sB[i * 8]);
  async_copies_done();

  int lane = tid & 31;
  int wv   = tid >> 5;             // column tile 0..3
  int n0   = wv * 16;
  int r    = lane & 15;
  int hi   = lane >> 4;
  float* Out = x_dbl + (long)bk * CDIM * LL + cc * XCHUNK;

  for (int tm = 0; tm < 3; ++tm) {
    v8f acc = {};
#pragma unroll
    for (int k0 = 0; k0 < DINNER; k0 += 32) {
      const v8h* pa = (const v8h*)&sA[(tm * 16 + r) * DINNER + k0 + hi * 8];
      v8h alo = pa[0];
      v8h ahi = pa[2];   // +16 halves
      const v8h* pb = (const v8h*)&sB[(n0 + r) * DINNER + k0 + hi * 8];
      v8h blo = pb[0];
      v8h bhi = pb[2];   // +16 halves
      v16h a, b;
#pragma unroll
      for (int e = 0; e < 8; ++e) {
        a[e] = alo[e]; a[8 + e] = ahi[e];
        b[e] = blo[e]; b[8 + e] = bhi[e];
      }
      acc = __builtin_amdgcn_wmma_f32_16x16x32_f16(false, a, false, b,
                                                   (short)0, acc, false, false);
    }
    int row0 = tm * 16 + hi * 8;
#pragma unroll
    for (int v = 0; v < 8; ++v) {
      int rr = row0 + v;
      if (rr < CDIM) Out[(long)rr * LL + n0 + r] = acc[v];
    }
  }
}

// ---------------- K5: delta = softplus(dt_w @ dts_rank + dt_b) ----------------
__global__ void k_delta(const float* __restrict__ x_dbl,
                        const float* __restrict__ dt_w,
                        const float* __restrict__ dt_b,
                        float* __restrict__ delta) {
  long idx = (long)blockIdx.x * blockDim.x + threadIdx.x;
  if (idx >= (long)BB * KK * DINNER * LL) return;
  int l  = (int)(idx & (LL - 1));
  int d  = (int)((idx >> 12) % DINNER);
  int bk = (int)(idx / ((long)DINNER * LL));
  int k  = bk & 1;
  const float* xd = x_dbl + (long)bk * CDIM * LL;
  float s = dt_b[k * DINNER + d];
#pragma unroll
  for (int r = 0; r < DTRANK; ++r)
    s += dt_w[((long)k * DINNER + d) * DTRANK + r] * xd[(long)r * LL + l];
  delta[idx] = (s > 20.f) ? s : log1pf(__expf(s));
}

// ---------------- K6: selective scan (serial over L) ----------------
__global__ void k_scan(const float* __restrict__ xs,
                       const float* __restrict__ x_dbl,
                       const float* __restrict__ delta,
                       const float* __restrict__ A_logs,
                       const float* __restrict__ Ds,
                       float* __restrict__ out_y) {
  __shared__ float sBC[64];
  int bk = blockIdx.x;
  int d  = threadIdx.x;          // 0..191
  int k  = bk & 1;
  const float* xd = x_dbl + (long)bk * CDIM * LL;
  const float* dl = delta + ((long)bk * DINNER + d) * LL;
  const float* uu = xs + ((long)bk * DINNER + d) * LL;
  float*       oy = out_y + ((long)bk * DINNER + d) * LL;
  float An[DSTATE];
#pragma unroll
  for (int n = 0; n < DSTATE; ++n)
    An[n] = -__expf(A_logs[((long)k * DINNER + d) * DSTATE + n]);
  float Dv = Ds[k * DINNER + d];
  float h[DSTATE];
#pragma unroll
  for (int n = 0; n < DSTATE; ++n) h[n] = 0.f;

  for (int l = 0; l < LL; ++l) {
    int par = (l & 1) * 32;
    if (threadIdx.x < 32)
      sBC[par + threadIdx.x] = xd[(long)(DTRANK + threadIdx.x) * LL + l];
    __syncthreads();
    float dv = dl[l];
    float u  = uu[l];
    float du = dv * u;
    float y = 0.f;
#pragma unroll
    for (int n = 0; n < DSTATE; ++n) {
      float a = __expf(dv * An[n]);
      h[n] = a * h[n] + du * sBC[par + n];
      y += h[n] * sBC[par + DSTATE + n];
    }
    oy[l] = y + Dv * u;
  }
}

// ---------------- K7: combine directions + inverse order ----------------
__global__ void k_combine(const float* __restrict__ out_y,
                          const int* __restrict__ inv,
                          float* __restrict__ ycomb) {
  long idx = (long)blockIdx.x * blockDim.x + threadIdx.x;
  if (idx >= (long)BB * LL * DINNER) return;
  int d = (int)(idx % DINNER);
  long t = idx / DINNER;
  int l = (int)(t & (LL - 1));
  int b = (int)(t >> 12);
  int ip = inv[l];
  float v = out_y[((long)(b * 2 + 0) * DINNER + d) * LL + ip] +
            out_y[((long)(b * 2 + 1) * DINNER + d) * LL + (LL - 1 - ip)];
  ycomb[idx] = v;
}

// ---------------- K8: layernorm(192) + SiLU(z) gate (in place) ----------------
__global__ void k_ln_gate(const float* __restrict__ xz,
                          const float* __restrict__ ln_g,
                          const float* __restrict__ ln_b,
                          float* __restrict__ ycomb) {
  __shared__ float r1[DINNER];
  __shared__ float r2[DINNER];
  long row = blockIdx.x;
  int d = threadIdx.x;
  float v = ycomb[row * DINNER + d];
  r1[d] = v;
  r2[d] = v * v;
  __syncthreads();
  for (int s = 96; s >= 3; s >>= 1) {
    if (d < s) { r1[d] += r1[d + s]; r2[d] += r2[d + s]; }
    __syncthreads();
  }
  if (d == 0) {
    r1[0] += r1[1] + r1[2];
    r2[0] += r2[1] + r2[2];
  }
  __syncthreads();
  float mu  = r1[0] * (1.f / DINNER);
  float var = r2[0] * (1.f / DINNER) - mu * mu;
  float yn = (v - mu) * rsqrtf(var + 1e-5f) * ln_g[d] + ln_b[d];
  float z = xz[row * (2 * DINNER) + DINNER + d];
  ycomb[row * DINNER + d] = yn * silu_f(z);
}

// ---------------- K9: output projection  out = gated @ w_out^T (WMMA) -------
__global__ void k_out_proj(const float* __restrict__ gated,
                           const float* __restrict__ w_out,
                           float* __restrict__ out) {
  const int N = DMODEL, Kd = DINNER;
  int wave = (blockIdx.x * blockDim.x + threadIdx.x) >> 5;
  int lane = threadIdx.x & 31;
  int tn = wave % (N / 16);   // 6
  int tm = wave / (N / 16);   // 512
  v8f c = {};
#pragma unroll
  for (int k0 = 0; k0 < Kd; k0 += 32) {
    v16h a = load_fragA(gated, Kd, 1, tm * 16, k0, lane);
    v16h b = load_fragB(w_out, 1, Kd, k0, tn * 16, lane);
    c = __builtin_amdgcn_wmma_f32_16x16x32_f16(false, a, false, b,
                                               (short)0, c, false, false);
  }
  int row0 = tm * 16 + (lane >> 4) * 8;
  int col  = tn * 16 + (lane & 15);
#pragma unroll
  for (int v = 0; v < 8; ++v)
    out[(long)(row0 + v) * N + col] = c[v];
}

// ---------------- host launcher ----------------
extern "C" void kernel_launch(void* const* d_in, const int* in_sizes, int n_in,
                              void* d_out, int out_size, void* d_ws, size_t ws_size,
                              hipStream_t stream) {
  const float* x        = (const float*)d_in[0];
  const float* w_in     = (const float*)d_in[1];
  const float* conv_w   = (const float*)d_in[2];
  const float* conv_b   = (const float*)d_in[3];
  const float* x_proj_w = (const float*)d_in[4];
  const float* dt_w     = (const float*)d_in[5];
  const float* dt_b     = (const float*)d_in[6];
  const float* A_logs   = (const float*)d_in[7];
  const float* Ds       = (const float*)d_in[8];
  const float* ln_g     = (const float*)d_in[9];
  const float* ln_b     = (const float*)d_in[10];
  const float* w_out    = (const float*)d_in[11];
  float* out = (float*)d_out;

  // workspace layout
  float* ws    = (float*)d_ws;
  float* xz    = ws;                                   // 8192*384
  float* xc    = xz + (long)MROWS * 2 * DINNER;        // 2*192*4096
  float* xs    = xc + (long)BB * DINNER * LL;          // 4*192*4096
  float* xdbl  = xs + (long)BB * KK * DINNER * LL;     // 4*38*4096
  float* delta = xdbl + (long)BB * KK * CDIM * LL;     // 4*192*4096
  float* outy  = delta + (long)BB * KK * DINNER * LL;  // 4*192*4096
  float* ycomb = delta;                                // alias: delta dead post-scan
  int*   order = (int*)(outy + (long)BB * KK * DINNER * LL);
  int*   inv   = order + LL;
  _Float16* xsh = (_Float16*)(inv + LL);               // 4*4096*192 f16 (blocked)
  _Float16* aph = xsh + (long)BB * KK * LL * DINNER;   // 2*48*192 f16 (padded)

  // K0: diagonal order tables
  k_order<<<(LL + 255) / 256, 256, 0, stream>>>(order, inv);

  // K1: in projection (512*24 waves -> 1536 blocks of 8 waves)
  k_in_proj<<<1536, 256, 0, stream>>>(x, w_in, xz);

  // K2: depthwise conv + SiLU
  {
    long n = (long)BB * DINNER * LL;
    k_conv<<<(unsigned)((n + 255) / 256), 256, 0, stream>>>(xz, conv_w, conv_b, xc);
  }

  // K3: build scan sequences (f32 + blocked f16)
  {
    long n = (long)BB * KK * DINNER * LL;
    k_build_xs<<<(unsigned)((n + 255) / 256), 256, 0, stream>>>(xc, order, xs, xsh);
  }

  // K3b: pack padded f16 A panels
  k_pack_aproj<<<(KK * CPAD * DINNER + 255) / 256, 256, 0, stream>>>(x_proj_w, aph);

  // K4: x_dbl projection (4 bk * 64 chunks -> 256 blocks of 128 threads)
  k_xdbl<<<256, 128, 0, stream>>>(xsh, aph, xdbl);

  // K5: delta
  {
    long n = (long)BB * KK * DINNER * LL;
    k_delta<<<(unsigned)((n + 255) / 256), 256, 0, stream>>>(xdbl, dt_w, dt_b, delta);
  }

  // K6: selective scan (4 sequences, 192 channels each)
  k_scan<<<BB * KK, DINNER, 0, stream>>>(xs, xdbl, delta, A_logs, Ds, outy);

  // K7: combine + un-permute
  {
    long n = (long)BB * LL * DINNER;
    k_combine<<<(unsigned)((n + 255) / 256), 256, 0, stream>>>(outy, inv, ycomb);
  }

  // K8: layernorm + gate (in place on ycomb)
  k_ln_gate<<<MROWS, DINNER, 0, stream>>>(xz, ln_g, ln_b, ycomb);

  // K9: out projection (512*6 waves -> 384 blocks)
  k_out_proj<<<384, 256, 0, stream>>>(ycomb, w_out, out);
}
